// MoleculeANI_69947837382786
// MI455X (gfx1250) — compile-verified, matrix operand order checked by reading, not run on hardware
//
#include <hip/hip_runtime.h>
#include <hip/hip_bf16.h>
#include <stdint.h>

typedef _Float16 half_t;
typedef __attribute__((ext_vector_type(16))) _Float16 v16h;
typedef __attribute__((ext_vector_type(8)))  float    v8f;
typedef __attribute__((ext_vector_type(4)))  float    v4f;
typedef __attribute__((ext_vector_type(4)))  int      v4i;

#define CELU_ALPHA 0.1f

// Problem constants (match the reference)
#define BB 512
#define NN 64
#define FF 768
#define TT 4
#define H1 128
#define H2 128
#define H3 64

// fragment tiling
#define KT1 24   // 768/32
#define NT1 8    // 128/16
#define KT2 4    // 128/32
#define NT2 8    // 128/16
#define KT3 4    // 128/32
#define NT3 4    // 64/16

// LDS layout (bytes)
#define ACT_BYTES   32768              // 8 waves x 16 rows x 128 f16
#define W1SLAB      8192               // one K-slab: 8 frags x 1KB
#define W1BUFS      3                  // triple buffer
#define SMEM_BYTES  (ACT_BYTES + W1BUFS * W1SLAB)   // 56 KB

union Frag {
    v16h h;
    v4i  i[2];
};

__device__ __forceinline__ float celu_f(float x) {
    return x > 0.0f ? x : CELU_ALPHA * (__expf(x * (1.0f / CELU_ALPHA)) - 1.0f);
}

// Issue one 32-byte chunk (2 x B128) of an async global->LDS copy.
// GV addressing: %0 = LDS byte offset (VGPR), %1 = 64-bit global address (VGPR pair).
__device__ __forceinline__ void issue_async32(const char* src, unsigned ldsOff) {
    uint64_t g = (uint64_t)(uintptr_t)src;
    asm volatile("global_load_async_to_lds_b128 %0, %1, off\n\t"
                 "global_load_async_to_lds_b128 %0, %1, off offset:16"
                 :: "v"(ldsOff), "v"(g)
                 : "memory");
}

// ---------------------------------------------------------------------------
// Pack f32 weights (T,K,N) into f16 WMMA B-fragments:
// fragment (s,kt,nt) is 32x16 f16, stored as 32 lanes x 8 dwords; lane l,
// dword j holds W[k, n],W[k+1, n] with
//   k = kt*32 + (l>>4)*8 + (j<4 ? 2j : 16+2(j-4)),  n = nt*16 + (l&15)
// so the main kernel's B load is two contiguous 16B loads per lane.
// ---------------------------------------------------------------------------
__device__ __forceinline__ void pack_one(const float* __restrict__ W,
                                         unsigned* __restrict__ dst,
                                         int idx, int ktiles, int ntiles, int Nn) {
    int j = idx & 7;
    int l = (idx >> 3) & 31;
    int t = idx >> 8;
    int nt = t % ntiles; t /= ntiles;
    int kt = t % ktiles;
    int s  = t / ktiles;
    int jj = (j < 4) ? (2 * j) : (16 + 2 * (j - 4));
    int k  = kt * 32 + (l >> 4) * 8 + jj;
    int n  = nt * 16 + (l & 15);
    int K  = ktiles * 32;
    const float* Ws = W + ((size_t)s * K + k) * Nn + n;
    union { half_t h[2]; unsigned u; } pk;
    pk.h[0] = (half_t)Ws[0];
    pk.h[1] = (half_t)Ws[(size_t)Nn];
    size_t fragIdx = ((size_t)s * ktiles + kt) * ntiles + nt;
    dst[(fragIdx * 32 + l) * 8 + j] = pk.u;
}

__global__ void ani_pack_weights(const float* __restrict__ W1,
                                 const float* __restrict__ W2,
                                 const float* __restrict__ W3,
                                 const float* __restrict__ W4,
                                 half_t* __restrict__ w1f,
                                 half_t* __restrict__ w2f,
                                 half_t* __restrict__ w3f,
                                 half_t* __restrict__ w4f) {
    const int C1 = TT * KT1 * NT1 * 256;   // 196608 dwords
    const int C2 = TT * KT2 * NT2 * 256;   //  32768 dwords
    const int C3 = TT * KT3 * NT3 * 256;   //  16384 dwords
    const int C4 = TT * H3;                //    256 halves
    int idx = blockIdx.x * blockDim.x + threadIdx.x;
    if (idx < C1) {
        pack_one(W1, (unsigned*)w1f, idx, KT1, NT1, H1);
    } else if (idx < C1 + C2) {
        pack_one(W2, (unsigned*)w2f, idx - C1, KT2, NT2, H2);
    } else if (idx < C1 + C2 + C3) {
        pack_one(W3, (unsigned*)w3f, idx - C1 - C2, KT3, NT3, H3);
    } else if (idx < C1 + C2 + C3 + C4) {
        int t = idx - C1 - C2 - C3;
        w4f[t] = (half_t)W4[t];
    }
}

__global__ void ani_zero_out(float* __restrict__ out) {
    int i = blockIdx.x * blockDim.x + threadIdx.x;
    if (i < BB) out[i] = 0.0f;
}

// ---------------------------------------------------------------------------
// Main fused kernel. grid = (64 atoms, 4 batch tiles); block = 256 = 8 waves.
// Wave w handles 16 batch rows: m0 = btile*128 + w*16. Per atom, all rows
// share the species-selected weights -> dense WMMA GEMM chain held in regs.
// Layer-1 B-slabs are triple-buffered into LDS via async global->LDS copies
// (ASYNCcnt), so each slab is fetched once per block instead of once per wave.
// ---------------------------------------------------------------------------
__global__ void __launch_bounds__(256)
ani_main(const float* __restrict__ data,
         const int*   __restrict__ species,
         const float* __restrict__ b1,
         const float* __restrict__ b2,
         const float* __restrict__ b3,
         const float* __restrict__ b4,
         const half_t* __restrict__ w1f,
         const half_t* __restrict__ w2f,
         const half_t* __restrict__ w3f,
         const half_t* __restrict__ w4f,
         float* __restrict__ out) {
    __shared__ char smem[SMEM_BYTES];
    half_t* act  = (half_t*)smem;                    // activation staging
    half_t* wbuf = (half_t*)(smem + ACT_BYTES);      // 3 x 8KB weight slabs

    const int atom  = blockIdx.x;          // 0..63
    const int btile = blockIdx.y;          // 0..3
    const int tid   = threadIdx.x;
    const int lane  = tid & 31;
    const int wave  = tid >> 5;            // 0..7
    const int m0    = btile * 128 + wave * 16;
    const int lrow  = lane & 15;
    const int lhi   = lane >> 4;           // 0 or 1
    const int s     = species[atom];

    half_t* myl = act + wave * 16 * 128;

    // ---------------- Layer 1: [16x768] @ [768x128] ----------------
    const float* xrow = data + ((size_t)(m0 + lrow) * NN + atom) * FF;
    const char*  w1bytes  = (const char*)w1f + (size_t)s * (KT1 * NT1 * 1024);
    const unsigned wbufBase = (unsigned)(uintptr_t)(smem + ACT_BYTES);

    // prologue: slabs 0 and 1 in flight (2 x b128 per thread per slab)
    issue_async32(w1bytes + 0 * W1SLAB + tid * 32, wbufBase + 0 * W1SLAB + tid * 32);
    issue_async32(w1bytes + 1 * W1SLAB + tid * 32, wbufBase + 1 * W1SLAB + tid * 32);

    v8f acc1[NT1] = {};
    for (int kt = 0; kt < KT1; ++kt) {
        // wait until this thread's portion of slab kt is in LDS (in-order
        // completion: <=2 outstanding means slab kt's 2 ops are done)
        if (kt < KT1 - 1) {
            asm volatile("s_wait_asynccnt 0x2" ::: "memory");
        } else {
            asm volatile("s_wait_asynccnt 0x0" ::: "memory");
        }
        __syncthreads();   // everyone's portion of slab kt landed; everyone
                           // also finished reading buffer (kt-1)%3 last iter
        if (kt + 2 < KT1) {
            // refill buffer (kt+2)%3 == (kt-1)%3, retired by the barrier above
            issue_async32(w1bytes + (size_t)(kt + 2) * W1SLAB + tid * 32,
                          wbufBase + ((kt + 2) % W1BUFS) * W1SLAB + tid * 32);
        }

        const int kb = kt * 32 + lhi * 8;
        // prefetch next K-slab of activations (gfx1250 global_prefetch_b8)
        if (kt + 1 < KT1) __builtin_prefetch(xrow + kb + 32, 0, 3);

        const v4f* p0 = (const v4f*)(xrow + kb);
        const v4f* p1 = (const v4f*)(xrow + kb + 16);
        v4f x0 = p0[0], x1 = p0[1], x2 = p1[0], x3 = p1[1];

        Frag A;
        #pragma unroll
        for (int i = 0; i < 4; ++i) {
            float a = x0[i]; A.h[i]      = (half_t)(a == a ? a : 0.0f);
            float b = x1[i]; A.h[4 + i]  = (half_t)(b == b ? b : 0.0f);
            float c = x2[i]; A.h[8 + i]  = (half_t)(c == c ? c : 0.0f);
            float d = x3[i]; A.h[12 + i] = (half_t)(d == d ? d : 0.0f);
        }

        const half_t* wl = wbuf + (kt % W1BUFS) * (W1SLAB / 2) + lane * 16;
        #pragma unroll
        for (int nt = 0; nt < NT1; ++nt) {
            Frag B;
            B.i[0] = *((const v4i*)(wl + nt * 512));
            B.i[1] = *((const v4i*)(wl + nt * 512 + 8));
            acc1[nt] = __builtin_amdgcn_wmma_f32_16x16x32_f16(
                false, A.h, false, B.h, (short)0, acc1[nt], false, false);
        }
    }

    // bias + CELU -> LDS (f16, row-major 16x128 per wave)
    #pragma unroll
    for (int nt = 0; nt < NT1; ++nt) {
        float bn = b1[s * H1 + nt * 16 + lrow];
        #pragma unroll
        for (int r = 0; r < 8; ++r) {
            float v = celu_f(acc1[nt][r] + bn);
            myl[(r + lhi * 8) * 128 + nt * 16 + lrow] = (half_t)v;
        }
    }

    // ---------------- Layer 2: [16x128] @ [128x128] ----------------
    Frag A2[KT2];
    #pragma unroll
    for (int kt = 0; kt < KT2; ++kt) {
        const int kb = kt * 32 + lhi * 8;
        A2[kt].i[0] = *((const v4i*)(myl + lrow * 128 + kb));
        A2[kt].i[1] = *((const v4i*)(myl + lrow * 128 + kb + 16));
    }

    const v4i* w2base = (const v4i*)w2f + (size_t)s * KT2 * NT2 * 64;
    v8f acc2[NT2] = {};
    #pragma unroll
    for (int kt = 0; kt < KT2; ++kt) {
        const v4i* wb = w2base + (size_t)kt * NT2 * 64 + lane * 2;
        #pragma unroll
        for (int nt = 0; nt < NT2; ++nt) {
            Frag B;
            B.i[0] = wb[nt * 64];
            B.i[1] = wb[nt * 64 + 1];
            acc2[nt] = __builtin_amdgcn_wmma_f32_16x16x32_f16(
                false, A2[kt].h, false, B.h, (short)0, acc2[nt], false, false);
        }
    }

    #pragma unroll
    for (int nt = 0; nt < NT2; ++nt) {
        float bn = b2[s * H2 + nt * 16 + lrow];
        #pragma unroll
        for (int r = 0; r < 8; ++r) {
            float v = celu_f(acc2[nt][r] + bn);
            myl[(r + lhi * 8) * 128 + nt * 16 + lrow] = (half_t)v;
        }
    }

    // ---------------- Layer 3: [16x128] @ [128x64] ----------------
    Frag A3[KT3];
    #pragma unroll
    for (int kt = 0; kt < KT3; ++kt) {
        const int kb = kt * 32 + lhi * 8;
        A3[kt].i[0] = *((const v4i*)(myl + lrow * 128 + kb));
        A3[kt].i[1] = *((const v4i*)(myl + lrow * 128 + kb + 16));
    }

    const v4i* w3base = (const v4i*)w3f + (size_t)s * KT3 * NT3 * 64;
    v8f acc3[NT3] = {};
    #pragma unroll
    for (int kt = 0; kt < KT3; ++kt) {
        const v4i* wb = w3base + (size_t)kt * NT3 * 64 + lane * 2;
        #pragma unroll
        for (int nt = 0; nt < NT3; ++nt) {
            Frag B;
            B.i[0] = wb[nt * 64];
            B.i[1] = wb[nt * 64 + 1];
            acc3[nt] = __builtin_amdgcn_wmma_f32_16x16x32_f16(
                false, A3[kt].h, false, B.h, (short)0, acc3[nt], false, false);
        }
    }

    // bias + CELU in registers
    #pragma unroll
    for (int nt = 0; nt < NT3; ++nt) {
        float bn = b3[s * H3 + nt * 16 + lrow];
        #pragma unroll
        for (int r = 0; r < 8; ++r) {
            acc3[nt][r] = celu_f(acc3[nt][r] + bn);
        }
    }

    // ---------------- Layer 4: [16x64] . [64x1] + b4, sum into out ----------
    float psum[8] = {};
    #pragma unroll
    for (int nt = 0; nt < NT3; ++nt) {
        float w = (float)w4f[s * H3 + nt * 16 + lrow];
        #pragma unroll
        for (int r = 0; r < 8; ++r) psum[r] += acc3[nt][r] * w;
    }

    const float bias4 = b4[s];
    #pragma unroll
    for (int r = 0; r < 8; ++r) {
        float v = psum[r];
        v += __shfl_xor(v, 8, 16);
        v += __shfl_xor(v, 4, 16);
        v += __shfl_xor(v, 2, 16);
        v += __shfl_xor(v, 1, 16);
        if (lrow == 0) {
            int row = m0 + lhi * 8 + r;
            atomicAdd(&out[row], v + bias4);
        }
    }
}

// ---------------------------------------------------------------------------
extern "C" void kernel_launch(void* const* d_in, const int* in_sizes, int n_in,
                              void* d_out, int out_size, void* d_ws, size_t ws_size,
                              hipStream_t stream) {
    (void)in_sizes; (void)n_in; (void)out_size; (void)ws_size;

    const float* data    = (const float*)d_in[0];
    const int*   species = (const int*)d_in[1];
    const float* W1 = (const float*)d_in[2];
    const float* b1 = (const float*)d_in[3];
    const float* W2 = (const float*)d_in[4];
    const float* b2 = (const float*)d_in[5];
    const float* W3 = (const float*)d_in[6];
    const float* b3 = (const float*)d_in[7];
    const float* W4 = (const float*)d_in[8];
    const float* b4 = (const float*)d_in[9];
    float* out = (float*)d_out;

    // workspace layout (f16 pre-swizzled weight fragments)
    char* ws = (char*)d_ws;
    half_t* w1f = (half_t*)(ws + 0);        // 786432 B
    half_t* w2f = (half_t*)(ws + 786432);   // 131072 B
    half_t* w3f = (half_t*)(ws + 917504);   //  65536 B
    half_t* w4f = (half_t*)(ws + 983040);   //    512 B

    const int packWork = TT * KT1 * NT1 * 256 + TT * KT2 * NT2 * 256 +
                         TT * KT3 * NT3 * 256 + TT * H3;   // 246016
    ani_pack_weights<<<(packWork + 255) / 256, 256, 0, stream>>>(
        W1, W2, W3, W4, w1f, w2f, w3f, w4f);

    ani_zero_out<<<(BB + 255) / 256, 256, 0, stream>>>(out);

    dim3 grid(NN, BB / 128);   // 64 atoms x 4 batch tiles
    ani_main<<<grid, 256, 0, stream>>>(data, species, b1, b2, b3, b4,
                                       w1f, w2f, w3f, w4f, out);
}
